// GPi_25434796327645
// MI455X (gfx1250) — matrix operand order repaired and checked by reading, not compiled, not run on hardware
//
#include <hip/hip_runtime.h>
#include <hip/hip_bf16.h>

// ---------------------------------------------------------------------------
// out[b,u] = sum_i x[b,i] * (mask[u,i] * w[i,u]) + bias[u]
//   == GEMM: X(8192x3072) @ Weff(3072x1536) + bias
// Strategy: split-bf16 ("bf16x3") WMMA GEMM with fp32 accumulation.
//   x = xh + xl, w = wh + wl (bf16 RNE halves)
//   x*w ~= xh*wh + xh*wl + xl*wh   (xl*wl ~ 2^-18 relative, dropped)
// 3x v_wmma_f32_16x16x32_bf16 per K=32 chunk vs 8x v_wmma_f32_16x16x4_f32:
// ~2.7x matrix-pipe throughput at ~1e-5 relative error.
// ---------------------------------------------------------------------------

typedef __bf16 bf16_t;
typedef __attribute__((ext_vector_type(16))) __bf16 v16bf;
typedef __attribute__((ext_vector_type(8)))  __bf16 v8bf;
typedef __attribute__((ext_vector_type(4)))  __bf16 v4bf;
typedef __attribute__((ext_vector_type(8)))  float  v8f;
typedef __attribute__((ext_vector_type(4)))  float  v4f;

#define BATCH 8192
#define KDIM  3072
#define NDIM  1536

#define BM 128
#define BN 128
#define BK 32
#define LDSP 40   // padded LDS row stride (elements) to avoid bank conflicts

// ---------------------------------------------------------------------------
// Prep: Weff[i,u] = w[i,u] * mask[u,i]; write transposed (N-major) split-bf16
// arrays Wh[u,i], Wl[u,i]. LDS-tiled 32x32 transpose so every global access
// is coalesced (w read coalesced over u, mask read + Wh/Wl write coalesced
// over i).
// ---------------------------------------------------------------------------
__global__ __launch_bounds__(256) void GPi_prep_split_w(
    const float* __restrict__ w, const float* __restrict__ mask,
    bf16_t* __restrict__ wh, bf16_t* __restrict__ wl)
{
    __shared__ float tile[32][33];
    const int i0 = blockIdx.x * 32;   // K (input) block
    const int u0 = blockIdx.y * 32;   // N (unit) block
    const int t  = threadIdx.x;

    // phase 1: read w[i, u0+tu] coalesced over u, stash in LDS
    {
        const int tu = t & 31;
        const int ti = t >> 5;        // 0..7
        #pragma unroll
        for (int r = 0; r < 4; ++r) {
            const int il = ti + r * 8;
            tile[il][tu] = w[(size_t)(i0 + il) * NDIM + (u0 + tu)];
        }
    }
    __syncthreads();
    // phase 2: apply mask (coalesced over i), split to bf16 hi/lo, write
    {
        const int pi = t & 31;
        const int pu = t >> 5;
        #pragma unroll
        for (int r = 0; r < 4; ++r) {
            const int ul = pu + r * 8;
            const int u  = u0 + ul;
            const int i  = i0 + pi;
            const float v = tile[pi][ul] * mask[(size_t)u * KDIM + i];
            const bf16_t h = (bf16_t)v;
            const bf16_t l = (bf16_t)(v - (float)h);
            const size_t o = (size_t)u * KDIM + i;
            wh[o] = h;
            wl[o] = l;
        }
    }
}

// ---------------------------------------------------------------------------
// Main GEMM: 128x128 C tile per block, 8 wave32 waves in a 2(M)x4(N) grid,
// each wave owns a 64x32 sub-tile = 4x2 WMMA 16x16 accumulators.
// K staged through LDS in 32-wide tiles; X converted fp32 -> (hi,lo) bf16 in
// registers during staging.
// ---------------------------------------------------------------------------
__global__ __launch_bounds__(256) void GPi_gemm_bf16x3(
    const float*  __restrict__ X,
    const bf16_t* __restrict__ Wh,
    const bf16_t* __restrict__ Wl,
    const float*  __restrict__ bias,
    float*        __restrict__ out)
{
    __shared__ __align__(16) bf16_t Ah[BM * LDSP];
    __shared__ __align__(16) bf16_t Al[BM * LDSP];
    __shared__ __align__(16) bf16_t Bh[BN * LDSP];
    __shared__ __align__(16) bf16_t Bl[BN * LDSP];

    const int t     = threadIdx.x;
    const int lane  = t & 31;
    const int wave  = t >> 5;
    const int waveM = wave >> 2;    // 0..1
    const int waveN = wave & 3;     // 0..3
    const int half  = lane >> 4;    // lane-half select (0 or 1)
    const int l16   = lane & 15;

    const int m0 = blockIdx.y * BM;
    const int n0 = blockIdx.x * BN;

    v8f acc[4][2];
    #pragma unroll
    for (int mt = 0; mt < 4; ++mt)
        #pragma unroll
        for (int nt = 0; nt < 2; ++nt)
            acc[mt][nt] = v8f{};

    for (int kt = 0; kt < KDIM / BK; ++kt) {
        const int k0 = kt * BK;

        // ---- stage A tile: 128x32 fp32 -> split bf16 (4 float4 / thread) ----
        #pragma unroll
        for (int j = 0; j < 4; ++j) {
            const int q   = t + j * 256;      // 1024 float4 chunks
            const int row = q >> 3;           // 8 float4 per row
            const int col = (q & 7) << 2;
            const v4f xv = *(const v4f*)(X + (size_t)(m0 + row) * KDIM + k0 + col);
            v4bf h, l;
            #pragma unroll
            for (int e = 0; e < 4; ++e) {
                const bf16_t hh = (bf16_t)xv[e];
                h[e] = hh;
                l[e] = (bf16_t)(xv[e] - (float)hh);
            }
            *(v4bf*)(Ah + row * LDSP + col) = h;
            *(v4bf*)(Al + row * LDSP + col) = l;
        }
        // ---- stage B tiles: 128x32 bf16 direct copies (2x16B / thread each) ----
        #pragma unroll
        for (int j = 0; j < 2; ++j) {
            const int c   = t + j * 256;      // 512 8-elem chunks
            const int row = c >> 2;           // 4 chunks per row
            const int col = (c & 3) << 3;
            *(v8bf*)(Bh + row * LDSP + col) =
                *(const v8bf*)(Wh + (size_t)(n0 + row) * KDIM + k0 + col);
            *(v8bf*)(Bl + row * LDSP + col) =
                *(const v8bf*)(Wl + (size_t)(n0 + row) * KDIM + k0 + col);
        }

        // prefetch next K tile into the cache hierarchy while we compute
        if (kt + 1 < KDIM / BK) {
            const int nk = k0 + BK;
            __builtin_prefetch(X + (size_t)(m0 + (t >> 1)) * KDIM + nk + (t & 1) * 16, 0, 3);
            if (t < 128) {
                __builtin_prefetch(Wh + (size_t)(n0 + t) * KDIM + nk, 0, 3);
                __builtin_prefetch(Wl + (size_t)(n0 + t) * KDIM + nk, 0, 3);
            }
        }
        __syncthreads();

        // ---- load fragments from LDS ----
        // A (16-bit 16x32, ISA layout): lanes 0-15: V0-3=K0..7, V4-7=K16..23;
        //                               lanes16-31: V0-3=K8..15, V4-7=K24..31.
        // B (32x16, N-major LDS): lanes 0-15: K0..7,K16..23 of column N=lane;
        //                         lanes16-31: K8..15,K24..31 of column N=lane-16.
        union Frag { v16bf v; v8bf h[2]; };
        Frag ah[4], al[4], bh[2], bl[2];
        #pragma unroll
        for (int mt = 0; mt < 4; ++mt) {
            const int row = waveM * 64 + mt * 16 + l16;
            const bf16_t* ph = Ah + row * LDSP;
            const bf16_t* pl = Al + row * LDSP;
            ah[mt].h[0] = *(const v8bf*)(ph + half * 8);
            ah[mt].h[1] = *(const v8bf*)(ph + 16 + half * 8);
            al[mt].h[0] = *(const v8bf*)(pl + half * 8);
            al[mt].h[1] = *(const v8bf*)(pl + 16 + half * 8);
        }
        #pragma unroll
        for (int nt = 0; nt < 2; ++nt) {
            const int row = waveN * 32 + nt * 16 + l16;
            const bf16_t* ph = Bh + row * LDSP;
            const bf16_t* pl = Bl + row * LDSP;
            bh[nt].h[0] = *(const v8bf*)(ph + half * 8);
            bh[nt].h[1] = *(const v8bf*)(ph + 16 + half * 8);
            bl[nt].h[0] = *(const v8bf*)(pl + half * 8);
            bl[nt].h[1] = *(const v8bf*)(pl + 16 + half * 8);
        }

        // ---- 24 WMMAs per wave per K-tile: hh + hl + lh ----
        #pragma unroll
        for (int mt = 0; mt < 4; ++mt) {
            #pragma unroll
            for (int nt = 0; nt < 2; ++nt) {
                v8f a = acc[mt][nt];
                a = __builtin_amdgcn_wmma_f32_16x16x32_bf16(
                        false, ah[mt].v, false, bh[nt].v, (short)0, a, false, false);
                a = __builtin_amdgcn_wmma_f32_16x16x32_bf16(
                        false, ah[mt].v, false, bl[nt].v, (short)0, a, false, false);
                a = __builtin_amdgcn_wmma_f32_16x16x32_bf16(
                        false, al[mt].v, false, bh[nt].v, (short)0, a, false, false);
                acc[mt][nt] = a;
            }
        }
        __syncthreads();
    }

    // ---- epilogue: C layout = VGPR r -> M=r (lanes 0-15) / M=r+8 (16-31) ----
    #pragma unroll
    for (int nt = 0; nt < 2; ++nt) {
        const int n  = n0 + waveN * 32 + nt * 16 + l16;
        const float bv = bias[n];
        #pragma unroll
        for (int mt = 0; mt < 4; ++mt) {
            const int mb = m0 + waveM * 64 + mt * 16 + half * 8;
            #pragma unroll
            for (int r = 0; r < 8; ++r) {
                out[(size_t)(mb + r) * NDIM + n] = acc[mt][nt][r] + bv;
            }
        }
    }
}

// ---------------------------------------------------------------------------
extern "C" void kernel_launch(void* const* d_in, const int* in_sizes, int n_in,
                              void* d_out, int out_size, void* d_ws, size_t ws_size,
                              hipStream_t stream)
{
    const float* x    = (const float*)d_in[0];   // (8192, 3072)
    const float* w    = (const float*)d_in[1];   // (3072, 1536)
    const float* b    = (const float*)d_in[2];   // (1536,)
    const float* mask = (const float*)d_in[3];   // (1536, 3072)
    float* out = (float*)d_out;                  // (8192, 1536)

    // workspace: transposed split weights, 2 * 3072*1536 bf16 = 18.9 MB
    bf16_t* wh = (bf16_t*)d_ws;
    bf16_t* wl = wh + (size_t)KDIM * NDIM;

    dim3 pgrid(KDIM / 32, NDIM / 32);
    hipLaunchKernelGGL(GPi_prep_split_w, pgrid, dim3(256), 0, stream, w, mask, wh, wl);

    dim3 ggrid(NDIM / BN, BATCH / BM);
    hipLaunchKernelGGL(GPi_gemm_bf16x3, ggrid, dim3(256), 0, stream, x, wh, wl, b, out);
}
